// FlaxRAR_75677323755912
// MI455X (gfx1250) — matrix operand order repaired and checked by reading, not compiled, not run on hardware
//
#include <hip/hip_runtime.h>

// ---------------- model dims ----------------
#define L_    12
#define C_    1024
#define H_    16
#define HD_   64
#define FF_   4096
#define S_    256
#define B_    16
#define CB_   1024
#define NTOK  257            // S + 1
#define M_    (B_ * NTOK)    // 4112 rows (tokens)
#define EPS_  1e-6f
#define KCHUNK 1024          // A-stripe staging chunk (16 x 1024 bf16 = 32KB LDS)

typedef __attribute__((ext_vector_type(16))) __bf16 v16bf;
typedef __attribute__((ext_vector_type(8)))  __bf16 v8bf;
typedef __attribute__((ext_vector_type(8)))  float  v8f;
typedef unsigned int uv4 __attribute__((ext_vector_type(4)));
typedef int          iv8 __attribute__((ext_vector_type(8)));
typedef int          iv4 __attribute__((ext_vector_type(4)));

#if defined(__gfx1250__) && __has_builtin(__builtin_amdgcn_tensor_load_to_lds)
#define HAVE_TDM 1
#else
#define HAVE_TDM 0
#endif

// ---------------- helpers ----------------
__device__ __forceinline__ float warp_sum32(float v) {
#pragma unroll
  for (int off = 16; off > 0; off >>= 1) v += __shfl_xor(v, off, 32);
  return v;
}

// ---------------- 1) embedding + conditioning ----------------
__global__ __launch_bounds__(256) void embed_kernel(
    const int* __restrict__ ids, const int* __restrict__ cond,
    const float* __restrict__ emb, const float* __restrict__ pos,
    float* __restrict__ x, float* __restrict__ cact_f, __bf16* __restrict__ cact_b) {
  int row = blockIdx.x;
  int b = row / NTOK, p = row % NTOK;
  int id = (p == 0) ? (cond[b] + CB_ + 1) : ids[b * S_ + (p - 1)];
  const float* e  = emb + (size_t)id * C_;
  const float* pp = pos + (size_t)p * C_;
  for (int c = threadIdx.x; c < C_; c += 256) {
    float v = e[c];
    x[(size_t)row * C_ + c] = v + pp[c];
    if (p == 0) {
      float a = v / (1.f + __expf(-v));    // silu
      cact_f[b * C_ + c] = a;
      cact_b[b * C_ + c] = (__bf16)a;
    }
  }
}

// ---------------- 2) weight fp32[K,Nc] -> bf16 transposed [Nc,K] -------------
__global__ __launch_bounds__(256) void transpose_cast_kernel(
    const float* __restrict__ W, __bf16* __restrict__ Wt, int K, int Nc) {
  __shared__ float tile[32][33];
  int tx = threadIdx.x & 31, ty = threadIdx.x >> 5;
  int nb = blockIdx.x * 32, kb = blockIdx.y * 32;
#pragma unroll
  for (int i = 0; i < 4; ++i)
    tile[ty + i * 8][tx] = W[(size_t)(kb + ty + i * 8) * Nc + nb + tx];
  __syncthreads();
#pragma unroll
  for (int i = 0; i < 4; ++i)
    Wt[(size_t)(nb + ty + i * 8) * K + kb + tx] = (__bf16)tile[tx][ty + i * 8];
}

// ---------------- 3) bf16 WMMA GEMM with TDM-staged A ------------------------
// out[M,Nc] = A[M,K] @ Wt[Nc,K]^T + bias; per wave 16x64 (1 A-frag x 4 B-frags),
// 8 waves/block -> 16 x 512 block tile. The shared 16xK A-stripe is DMA'd into
// LDS by the Tensor Data Mover in 16x1024 chunks (one tensor_load_to_lds per
// chunk, s_wait_tensorcnt + barrier), then all 8 waves read A fragments via
// ds_load_b128 while the per-wave B stream comes from global (L2-resident).
// mode 0: f32 out = acc+bias
// mode 1: f32 out = resid + gate[b,col]*(acc+bias)   (in-place residual ok)
// mode 2: bf16 out = gelu_tanh(acc+bias)
__global__ __launch_bounds__(256) void gemm_bf16_kernel(
    const __bf16* __restrict__ A, const __bf16* __restrict__ Wt,
    const float* __restrict__ bias,
    float* __restrict__ out_f32, __bf16* __restrict__ out_bf16,
    const float* __restrict__ resid, const float* __restrict__ gate,
    int gateStride, int gateOff, int K, int Nc, int mode) {
  __shared__ __bf16 sA[16 * KCHUNK];                 // 32 KB
  int wave = threadIdx.x >> 5;
  int lane = threadIdx.x & 31;
  int l  = lane & 15;
  int hi = lane >> 4;                 // lane half selects K sub-range
  int tileM = blockIdx.y * 16;
  int nBase = blockIdx.x * 512 + wave * 64;

  v8f acc[4] = {};

  for (int kc = 0; kc < K; kc += KCHUNK) {
#if HAVE_TDM
    if (wave == 0) {
      // ---- Tensor DMA descriptor (ISA 8.3-8.6): 2D tile 16 rows x 1024 cols
      unsigned long long ga =
          (unsigned long long)(uintptr_t)(A + (size_t)tileM * K + kc);
      unsigned ldsOff = (unsigned)(uintptr_t)(void*)sA;  // low 32b = LDS byte addr
      uv4 g0{};
      g0[0] = 1u;                                   // count=1, user descriptor
      g0[1] = ldsOff;                               // lds_addr
      g0[2] = (unsigned)(ga & 0xffffffffu);         // global_addr[31:0]
      g0[3] = (unsigned)((ga >> 32) & 0x01ffffffu)  // global_addr[56:32]
              | (2u << 30);                         // type = 2 ("image")
      iv8 g1{};
      g1[0] = 0x00010000;                           // wg_mask=0, data_size=1 (2B)
      g1[1] = (K & 0xffff) << 16;                   // tensor_dim0 low16 @bit48
      g1[2] = ((unsigned)K >> 16) | (16u << 16);    // tensor_dim0 hi | tensor_dim1=16
      g1[3] = (KCHUNK << 16);                       // tile_dim0 = 1024
      g1[4] = 16;                                   // tile_dim1 = 16
      g1[5] = K;                                    // tensor_dim0_stride low32
      g1[6] = 0;
      g1[7] = 0;
      iv4 g2{}, g3{};                               // dims 2..4 unused (zero)
      iv8 g4{};                                     // trailing group (unused), zero
      __builtin_amdgcn_tensor_load_to_lds(g0, g1, g2, g3, g4, 0);
      __builtin_amdgcn_s_wait_tensorcnt(0);
    }
#else
    // cooperative fallback: 256 threads copy 16x1024 bf16 (2048 x 16B chunks)
    for (int it = 0; it < 8; ++it) {
      int idx = threadIdx.x + it * 256;             // 0..2047
      int r = idx >> 6, ck = idx & 63;
      *(v8bf*)(sA + r * KCHUNK + ck * 8) =
          *(const v8bf*)(A + (size_t)(tileM + r) * K + kc + ck * 8);
    }
#endif
    __syncthreads();

    for (int k0 = 0; k0 < KCHUNK; k0 += 32) {
      // A 16x32 frag from LDS: lanes 0-15 K{0..15}, 16-31 K{16..31}
      const __bf16* ar = sA + l * KCHUNK + k0 + (hi << 4);
      v8bf alo = *(const v8bf*)(ar);
      v8bf ahi = *(const v8bf*)(ar + 8);
      v16bf afrag = __builtin_shufflevector(alo, ahi, 0, 1, 2, 3, 4, 5, 6, 7,
                                            8, 9, 10, 11, 12, 13, 14, 15);
      const __bf16* wbase = Wt + (size_t)(kc + k0) + (hi << 4);
#pragma unroll
      for (int f = 0; f < 4; ++f) {
        // B 32x16 frag: lane = column, lanes 0-15 K=0..15, lanes 16-31 K=16..31
        v16bf bfrag = *(const v16bf*)(wbase + (size_t)(nBase + f * 16 + l) * K);
        acc[f] = __builtin_amdgcn_wmma_f32_16x16x32_bf16(
            false, afrag, false, bfrag, (short)0, acc[f], false, false);
      }
    }
    __syncthreads();   // protect sA before next chunk's DMA overwrites it
  }

  // D layout: lanes 0-15 -> rows tileM+0..7, lanes 16-31 -> rows tileM+8..15
  int row0 = tileM + (hi << 3);
#pragma unroll
  for (int f = 0; f < 4; ++f) {
    int col = nBase + f * 16 + l;
    float bc = bias[col];
#pragma unroll
    for (int r = 0; r < 8; ++r) {
      int mm = row0 + r;
      float v = acc[f][r] + bc;
      size_t oidx = (size_t)mm * Nc + col;
      if (mode == 0) {
        out_f32[oidx] = v;
      } else if (mode == 1) {
        float g = gate[(size_t)(mm / NTOK) * gateStride + gateOff + col];
        out_f32[oidx] = resid[oidx] + g * v;
      } else {
        float t = tanhf(0.7978845608028654f * (v + 0.044715f * v * v * v));
        out_bf16[oidx] = (__bf16)(0.5f * v * (1.f + t));
      }
    }
  }
}

// ---------------- 4) LayerNorm + adaLN modulate, write bf16 ------------------
__global__ __launch_bounds__(256) void ln_mod_kernel(
    const float* __restrict__ x, const float* __restrict__ s,
    const float* __restrict__ bb, const float* __restrict__ mod,
    int modStride, int shOff, int scOff, __bf16* __restrict__ out) {
  int row = blockIdx.x;
  int b = row / NTOK;
  const float* xr = x + (size_t)row * C_;
  float ls = 0.f, lq = 0.f;
  for (int c = threadIdx.x; c < C_; c += 256) {
    float v = xr[c];
    ls += v; lq += v * v;
  }
  ls = warp_sum32(ls); lq = warp_sum32(lq);
  __shared__ float s1[8], s2[8];
  int wave = threadIdx.x >> 5, lane = threadIdx.x & 31;
  if (lane == 0) { s1[wave] = ls; s2[wave] = lq; }
  __syncthreads();
  float tot = 0.f, tq = 0.f;
#pragma unroll
  for (int i = 0; i < 8; ++i) { tot += s1[i]; tq += s2[i]; }
  float mu = tot * (1.f / C_);
  float var = tq * (1.f / C_) - mu * mu;
  float rinv = rsqrtf(var + EPS_);
  const float* mb = mod + (size_t)b * modStride;
  for (int c = threadIdx.x; c < C_; c += 256) {
    float y = (xr[c] - mu) * rinv;
    if (s) y = y * s[c] + bb[c];
    y = y * (1.f + mb[scOff + c]) + mb[shOff + c];
    out[(size_t)row * C_ + c] = (__bf16)y;
  }
}

// ---------------- 5) per-head q/k LayerNorm (HD=64) --------------------------
__global__ __launch_bounds__(256) void qk_ln_kernel(
    const float* __restrict__ qkv, const float* __restrict__ qs,
    const float* __restrict__ qb, const float* __restrict__ ks,
    const float* __restrict__ kb, float* __restrict__ Q,
    float* __restrict__ Ko, float* __restrict__ V) {
  int idx = blockIdx.x * 8 + (threadIdx.x >> 5);   // over B*H*N
  int lane = threadIdx.x & 31;
  int n = idx % NTOK;
  int bh = idx / NTOK;
  int h = bh % H_, b = bh / H_;
  size_t mrow = (size_t)(b * NTOK + n) * (3 * C_) + h * HD_;
  const float* qp = qkv + mrow;
  const float* kp = qkv + mrow + C_;
  const float* vp = qkv + mrow + 2 * C_;
  float q0 = qp[lane], q1 = qp[lane + 32];
  float k0 = kp[lane], k1 = kp[lane + 32];
  float sq = warp_sum32(q0 + q1), sqq = warp_sum32(q0 * q0 + q1 * q1);
  float sk = warp_sum32(k0 + k1), skk = warp_sum32(k0 * k0 + k1 * k1);
  float qmu = sq * (1.f / HD_), qrv = rsqrtf(sqq * (1.f / HD_) - qmu * qmu + EPS_);
  float kmu = sk * (1.f / HD_), krv = rsqrtf(skk * (1.f / HD_) - kmu * kmu + EPS_);
  size_t o = (size_t)bh * NTOK * HD_ + (size_t)n * HD_;
  Q[o + lane]       = (q0 - qmu) * qrv * qs[lane]      + qb[lane];
  Q[o + lane + 32]  = (q1 - qmu) * qrv * qs[lane + 32] + qb[lane + 32];
  Ko[o + lane]      = (k0 - kmu) * krv * ks[lane]      + kb[lane];
  Ko[o + lane + 32] = (k1 - kmu) * krv * ks[lane + 32] + kb[lane + 32];
  V[o + lane] = vp[lane];
  V[o + lane + 32] = vp[lane + 32];
}

// ---------------- 6) causal attention, online softmax ------------------------
__global__ __launch_bounds__(256) void attn_kernel(
    const float* __restrict__ Q, const float* __restrict__ K,
    const float* __restrict__ V, __bf16* __restrict__ O) {
  int bh = blockIdx.x;
  int b = bh / H_, h = bh % H_;
  const float* Qb = Q + (size_t)bh * NTOK * HD_;
  const float* Kb = K + (size_t)bh * NTOK * HD_;
  const float* Vb = V + (size_t)bh * NTOK * HD_;
  for (int qi = threadIdx.x; qi < NTOK; qi += 256) {
    float qv[HD_];
#pragma unroll
    for (int d = 0; d < HD_; ++d) qv[d] = Qb[(size_t)qi * HD_ + d] * 0.125f;
    float mmax = -3.4e38f, lsum = 0.f;
    float acc[HD_];
#pragma unroll
    for (int d = 0; d < HD_; ++d) acc[d] = 0.f;
    for (int j = 0; j <= qi; ++j) {
      const float* kr = Kb + (size_t)j * HD_;
      float sdot = 0.f;
#pragma unroll
      for (int d = 0; d < HD_; ++d) sdot += qv[d] * kr[d];
      float mn = fmaxf(mmax, sdot);
      float corr = __expf(mmax - mn);
      float p = __expf(sdot - mn);
      lsum = lsum * corr + p;
      const float* vr = Vb + (size_t)j * HD_;
#pragma unroll
      for (int d = 0; d < HD_; ++d) acc[d] = acc[d] * corr + p * vr[d];
      mmax = mn;
    }
    float inv = 1.f / lsum;
    __bf16* orow = O + (size_t)(b * NTOK + qi) * C_ + h * HD_;
#pragma unroll
    for (int d = 0; d < HD_; ++d) orow[d] = (__bf16)(acc[d] * inv);
  }
}

// ---------------- host: orchestration ----------------
extern "C" void kernel_launch(void* const* d_in, const int* in_sizes, int n_in,
                              void* d_out, int out_size, void* d_ws, size_t ws_size,
                              hipStream_t stream) {
  (void)in_sizes; (void)n_in; (void)out_size; (void)ws_size;
  const int*   input_ids = (const int*)d_in[0];
  const int*   condition = (const int*)d_in[1];
  const float* embed_t   = (const float*)d_in[2];
  const float* pos_e     = (const float*)d_in[3];
  const float* n1s = (const float*)d_in[4];
  const float* n1b = (const float*)d_in[5];
  const float* qkv_w = (const float*)d_in[6];
  const float* qkv_b = (const float*)d_in[7];
  const float* qn_s = (const float*)d_in[8];
  const float* qn_b = (const float*)d_in[9];
  const float* kn_s = (const float*)d_in[10];
  const float* kn_b = (const float*)d_in[11];
  const float* proj_w = (const float*)d_in[12];
  const float* proj_b = (const float*)d_in[13];
  const float* n2s = (const float*)d_in[14];
  const float* n2b = (const float*)d_in[15];
  const float* fc1_w = (const float*)d_in[16];
  const float* fc1_b = (const float*)d_in[17];
  const float* fc2_w = (const float*)d_in[18];
  const float* fc2_b = (const float*)d_in[19];
  const float* ada_w = (const float*)d_in[20];
  const float* ada_b = (const float*)d_in[21];
  const float* fin_w = (const float*)d_in[22];
  const float* fin_b = (const float*)d_in[23];
  const float* head_w = (const float*)d_in[24];
  const float* head_b = (const float*)d_in[25];

  // workspace carve-up (stream-ordered reuse of wt buffer per layer)
  char* ws = (char*)d_ws;
  size_t off = 0;
  auto carve = [&](size_t bytes) {
    off = (off + 255) & ~(size_t)255;
    char* p = ws + off;
    off += bytes;
    return p;
  };
  __bf16* wt    = (__bf16*)carve((size_t)6144 * 1024 * 2);   // max transposed weight
  float*  x     = (float*) carve((size_t)M_ * C_ * 4);
  __bf16* hbuf  = (__bf16*)carve((size_t)M_ * C_ * 2);
  float*  qkvo  = (float*) carve((size_t)M_ * 3 * C_ * 4);
  float*  Qb    = (float*) carve((size_t)B_ * H_ * NTOK * HD_ * 4);
  float*  Kb    = (float*) carve((size_t)B_ * H_ * NTOK * HD_ * 4);
  float*  Vb    = (float*) carve((size_t)B_ * H_ * NTOK * HD_ * 4);
  __bf16* Obuf  = (__bf16*)carve((size_t)M_ * C_ * 2);
  __bf16* m1    = (__bf16*)carve((size_t)M_ * FF_ * 2);
  float*  modb  = (float*) carve((size_t)B_ * 6 * C_ * 4);
  float*  fm    = (float*) carve((size_t)B_ * 2 * C_ * 4);
  float*  cactf = (float*) carve((size_t)B_ * C_ * 4);
  __bf16* cactb = (__bf16*)carve((size_t)B_ * C_ * 2);
  __bf16* xf    = (__bf16*)carve((size_t)M_ * C_ * 2);

  dim3 blk(256);

  embed_kernel<<<M_, blk, 0, stream>>>(input_ids, condition, embed_t, pos_e,
                                       x, cactf, cactb);

  for (int l = 0; l < L_; ++l) {
    // adaLN modulation: mod[B,6C] = silu(c) @ aw + ab
    transpose_cast_kernel<<<dim3(6 * C_ / 32, C_ / 32), blk, 0, stream>>>(
        ada_w + (size_t)l * C_ * 6 * C_, wt, C_, 6 * C_);
    gemm_bf16_kernel<<<dim3(6 * C_ / 512, 1), blk, 0, stream>>>(
        cactb, wt, ada_b + (size_t)l * 6 * C_, modb, nullptr, nullptr, nullptr,
        0, 0, C_, 6 * C_, 0);

    // MSA: h = LN1(x)*(1+sc_msa)+sh_msa  (shOff=0, scOff=C)
    ln_mod_kernel<<<M_, blk, 0, stream>>>(x, n1s + (size_t)l * C_,
                                          n1b + (size_t)l * C_, modb, 6 * C_,
                                          0, C_, hbuf);
    // qkv GEMM
    transpose_cast_kernel<<<dim3(3 * C_ / 32, C_ / 32), blk, 0, stream>>>(
        qkv_w + (size_t)l * C_ * 3 * C_, wt, C_, 3 * C_);
    gemm_bf16_kernel<<<dim3(3 * C_ / 512, M_ / 16), blk, 0, stream>>>(
        hbuf, wt, qkv_b + (size_t)l * 3 * C_, qkvo, nullptr, nullptr, nullptr,
        0, 0, C_, 3 * C_, 0);
    // per-head q/k LN + scatter
    qk_ln_kernel<<<(B_ * H_ * NTOK) / 8, blk, 0, stream>>>(
        qkvo, qn_s + (size_t)l * HD_, qn_b + (size_t)l * HD_,
        kn_s + (size_t)l * HD_, kn_b + (size_t)l * HD_, Qb, Kb, Vb);
    // causal attention
    attn_kernel<<<B_ * H_, blk, 0, stream>>>(Qb, Kb, Vb, Obuf);
    // proj GEMM + gate*out + residual (in place on x), gateOff = 2C (g_msa)
    transpose_cast_kernel<<<dim3(C_ / 32, C_ / 32), blk, 0, stream>>>(
        proj_w + (size_t)l * C_ * C_, wt, C_, C_);
    gemm_bf16_kernel<<<dim3(C_ / 512, M_ / 16), blk, 0, stream>>>(
        Obuf, wt, proj_b + (size_t)l * C_, x, nullptr, x, modb, 6 * C_,
        2 * C_, C_, C_, 1);

    // MLP: h = LN2(x)*(1+sc_mlp)+sh_mlp  (shOff=3C, scOff=4C)
    ln_mod_kernel<<<M_, blk, 0, stream>>>(x, n2s + (size_t)l * C_,
                                          n2b + (size_t)l * C_, modb, 6 * C_,
                                          3 * C_, 4 * C_, hbuf);
    transpose_cast_kernel<<<dim3(FF_ / 32, C_ / 32), blk, 0, stream>>>(
        fc1_w + (size_t)l * C_ * FF_, wt, C_, FF_);
    gemm_bf16_kernel<<<dim3(FF_ / 512, M_ / 16), blk, 0, stream>>>(
        hbuf, wt, fc1_b + (size_t)l * FF_, nullptr, m1, nullptr, nullptr,
        0, 0, C_, FF_, 2);   // gelu -> bf16
    transpose_cast_kernel<<<dim3(C_ / 32, FF_ / 32), blk, 0, stream>>>(
        fc2_w + (size_t)l * FF_ * C_, wt, FF_, C_);
    gemm_bf16_kernel<<<dim3(C_ / 512, M_ / 16), blk, 0, stream>>>(
        m1, wt, fc2_b + (size_t)l * C_, x, nullptr, x, modb, 6 * C_,
        5 * C_, FF_, C_, 1); // gateOff = 5C (g_mlp)
  }

  // final modulation: fm[B,2C] = silu(c) @ final_w + final_b
  transpose_cast_kernel<<<dim3(2 * C_ / 32, C_ / 32), blk, 0, stream>>>(
      fin_w, wt, C_, 2 * C_);
  gemm_bf16_kernel<<<dim3(2 * C_ / 512, 1), blk, 0, stream>>>(
      cactb, wt, fin_b, fm, nullptr, nullptr, nullptr, 0, 0, C_, 2 * C_, 0);
  // final LN (no affine): scale at offset 0, shift at offset C
  ln_mod_kernel<<<M_, blk, 0, stream>>>(x, nullptr, nullptr, fm, 2 * C_,
                                        C_, 0, xf);
  // head GEMM -> logits f32 [B,N,CB] directly into d_out
  transpose_cast_kernel<<<dim3(CB_ / 32, C_ / 32), blk, 0, stream>>>(
      head_w, wt, C_, CB_);
  gemm_bf16_kernel<<<dim3(CB_ / 512, M_ / 16), blk, 0, stream>>>(
      xf, wt, head_b, (float*)d_out, nullptr, nullptr, nullptr, 0, 0,
      C_, CB_, 0);
}
